// DualGCNClassifier_46729244180946
// MI455X (gfx1250) — compile-verified, matrix operand order checked by reading, not run on hardware
//
#include <hip/hip_runtime.h>
#include <math.h>
#include <stdint.h>

#define B_   64
#define S_   512
#define DINP 384          // 360 padded to 384 (mult of 32)
#define H_   256
#define H4   1024
#define D_   512
#define NH_  4
#define MEM_ 256
#define NBIN 4096

typedef _Float16 f16;
typedef __attribute__((ext_vector_type(8)))  _Float16 v8h;
typedef __attribute__((ext_vector_type(16))) _Float16 v16h;
typedef __attribute__((ext_vector_type(8)))  float    v8f;
typedef int v4i_t __attribute__((ext_vector_type(4)));
typedef int v8i_t __attribute__((ext_vector_type(8)));

// ---- WMMA fragment load: row-major 32-half row -> v16h per ISA 7.12.2 ----
// lane L: row = L&15 ; halves 0-7 = K[(L>>4)*8 .. +7], halves 8-15 = K[16+(L>>4)*8 .. +7]
__device__ __forceinline__ v16h load_frag(const f16* row, int kg) {
  v8h lo = *(const v8h*)(row + kg);
  v8h hi = *(const v8h*)(row + 16 + kg);
  return __builtin_shufflevector(lo, hi, 0,1,2,3,4,5,6,7,8,9,10,11,12,13,14,15);
}

// ---- CDNA5 async global->LDS copy (ASYNCcnt tracked, no VGPR round trip) ----
__device__ __forceinline__ void async_ld_b128(void* lds, const void* g) {
  unsigned l = (unsigned)(uintptr_t)lds;   // generic LDS ptr low 32 bits = LDS offset
  asm volatile("global_load_async_to_lds_b128 %0, %1, off"
               :: "v"(l), "v"(g) : "memory");
}
__device__ __forceinline__ void wait_async0() {
  asm volatile("s_wait_asynccnt 0x0" ::: "memory");
}

// ---- TDM: DMA 16 rows x 1024 f32 (row stride S_*H4 elems) into LDS ----
// D# per CDNA5 ISA (2D tile, data_size=4B, type=2)
__device__ __forceinline__ void tdm_load_16rows(const float* gsrc, const float* ldsdst) {
  unsigned long long ga = (unsigned long long)(uintptr_t)gsrc;
  unsigned lds = (unsigned)(uintptr_t)ldsdst;
  v4i_t g0;
  g0[0] = 1;                                               // count=1 (user descriptor)
  g0[1] = (int)lds;                                        // lds_addr
  g0[2] = (int)(unsigned)(ga & 0xffffffffull);             // global_addr[31:0]
  g0[3] = (int)((unsigned)((ga >> 32) & 0x01ffffffull) | (2u << 30)); // addr[56:32]|type=2
  v8i_t g1;
  g1[0] = 0x00020000;                                      // data_size=2 (4 bytes)
  g1[1] = (int)((H4 & 0xffff) << 16);                      // tensor_dim0[15:0]
  g1[2] = (int)((H4 >> 16) | (16 << 16));                  // tensor_dim0[31:16]|tensor_dim1[15:0]
  g1[3] = (int)(H4 << 16);                                 // tensor_dim1[31:16]=0 | tile_dim0
  g1[4] = 16;                                              // tile_dim1 = 16 rows
  g1[5] = (int)(S_ * H4);                                  // tensor_dim0_stride[31:0]
  g1[6] = 0;                                               // stride hi / dim1_stride lo
  g1[7] = 0;
  asm volatile("tensor_load_to_lds %0, %1" :: "s"(g0), "s"(g1) : "memory");
}

__device__ __forceinline__ float blk_reduce(float v, float* sbuf, int ismax) {
#pragma unroll
  for (int off = 16; off > 0; off >>= 1) {
    float o = __shfl_xor(v, off, 32);
    v = ismax ? fmaxf(v, o) : (v + o);
  }
  if ((threadIdx.x & 31) == 0) sbuf[threadIdx.x >> 5] = v;
  __syncthreads();
  float r = sbuf[0];
#pragma unroll
  for (int i = 1; i < 8; ++i) r = ismax ? fmaxf(r, sbuf[i]) : (r + sbuf[i]);
  __syncthreads();
  return r;
}

__device__ __forceinline__ float sigf(float x) { return 1.f / (1.f + __expf(-x)); }

// ===================== generic batched WMMA GEMM =====================
// C[M,N] = A[M,K] * op(B) (+bias)(/denom)(relu); op(B)=B[N,K]^T if bTrans
// else B is [K,N] (staged transposed into LDS). ldc == N always.
__global__ __launch_bounds__(256) void gemm_wmma_kernel(
    const f16* __restrict__ A, const f16* __restrict__ Bm,
    float* __restrict__ Cf, f16* __restrict__ Ch,
    const float* __restrict__ bias, const float* __restrict__ denom,
    int M, int N, int K, int lda, int ldb,
    int zdiv, long long sA1, long long sA2, long long sB1, long long sB2,
    long long sC1, long long sC2, int bTrans, int relu)
{
  __shared__ f16 As[64 * 32];
  __shared__ f16 Bs[64 * 32];

  {
    int z = blockIdx.z;
    long long zq = z / zdiv, zr = z % zdiv;
    A  += zq * sA1 + zr * sA2;
    Bm += zq * sB1 + zr * sB2;
    long long coff = zq * sC1 + zr * sC2;
    if (Cf) Cf += coff;
    if (Ch) Ch += coff;
  }
  const int m0 = blockIdx.y * 64;
  const int n0 = blockIdx.x * 64;
  const int t = threadIdx.x;
  const int w = t >> 5, lane = t & 31;
  const int wn = (w & 3) * 16;      // N sub-tile
  const int wm = (w >> 2) * 16;     // M sub-tiles at wm and wm+32
  const int row16 = lane & 15;
  const int kg = (lane >> 4) * 8;

  v8f acc0 = {}; v8f acc1 = {};

  for (int k0 = 0; k0 < K; k0 += 32) {
    { // stage A 64x32 via async global->LDS DMA
      int r = t >> 2, c = (t & 3) * 8;
      async_ld_b128(&As[r * 32 + c], &A[(long long)(m0 + r) * lda + k0 + c]);
      if (k0 + 32 < K)
        __builtin_prefetch(&A[(long long)(m0 + r) * lda + k0 + 32 + c], 0, 1);
    }
    if (bTrans) { // B is [N,K] row-major: async stage straight
      int r = t >> 2, c = (t & 3) * 8;
      async_ld_b128(&Bs[r * 32 + c], &Bm[(long long)(n0 + r) * ldb + k0 + c]);
      if (k0 + 32 < K)
        __builtin_prefetch(&Bm[(long long)(n0 + r) * ldb + k0 + 32 + c], 0, 1);
    } else {      // B is [K,N] row-major: stage with transpose -> [N][32]
      int kk = t >> 3, n8 = (t & 7) * 8;
      v8h v = *(const v8h*)&Bm[(long long)(k0 + kk) * ldb + n0 + n8];
#pragma unroll
      for (int i = 0; i < 8; ++i) Bs[(n8 + i) * 32 + kk] = v[i];
      if (k0 + 32 < K)
        __builtin_prefetch(&Bm[(long long)(k0 + 32 + kk) * ldb + n0 + n8], 0, 1);
    }
    wait_async0();
    __syncthreads();

    v16h a0 = load_frag(&As[(wm + row16) * 32], kg);
    v16h a1 = load_frag(&As[(wm + 32 + row16) * 32], kg);
    v16h bf = load_frag(&Bs[(wn + row16) * 32], kg);
    acc0 = __builtin_amdgcn_wmma_f32_16x16x32_f16(false, a0, false, bf, (short)0, acc0, false, false);
    acc1 = __builtin_amdgcn_wmma_f32_16x16x32_f16(false, a1, false, bf, (short)0, acc1, false, false);
    __syncthreads();
  }

  // epilogue: C layout lane(n)=wn+(lane&15), row=wm + r + (lane>>4)*8 (+32 for acc1)
  const int nl = n0 + wn + (lane & 15);
#pragma unroll
  for (int r = 0; r < 8; ++r) {
    int ml0 = m0 + wm + r + ((lane >> 4) << 3);
#pragma unroll
    for (int sub = 0; sub < 2; ++sub) {
      int gm = ml0 + sub * 32;
      float v = sub ? acc1[r] : acc0[r];
      if (bias)  v += bias[nl];
      if (denom) v /= denom[gm];
      if (relu)  v = fmaxf(v, 0.f);
      long long ci = (long long)gm * N + nl;
      if (Cf) Cf[ci] = v;
      if (Ch) Ch[ci] = (f16)v;
    }
  }
}

// ===================== persistent LSTM recurrence =====================
// 8 blocks: dir(2) x batch-group(4, 16 rows each). h/c in LDS, per-step
// gates = h @ Whh^T via WMMA; next step's xproj slice (64KB) is TDM-prefetched
// into a double-buffered LDS staging area, overlapped with the gate GEMM.
#define XBUF_ELEMS (16 * 1024)
#define LSTM_SMEM (16*256*2 + 16*256*4 + 16*1024*4 + 2*XBUF_ELEMS*4)   // 221184 B
__global__ __launch_bounds__(256) void lstm_rec_kernel(
    const float* __restrict__ xprojF, const float* __restrict__ xprojB,
    const f16* __restrict__ WhhFh, const f16* __restrict__ WhhBh,
    float* __restrict__ hseqF, float* __restrict__ hseqB)
{
  extern __shared__ char smem[];
  f16*   hs   = (f16*)smem;                                   // [16][256]
  float* cs   = (float*)(smem + 16 * 256 * 2);                // [16][256]
  float* gs   = (float*)(smem + 16 * 256 * 2 + 16 * 256 * 4); // [16][1024]
  float* xbuf = (float*)(smem + 16 * 256 * 2 + 16 * 256 * 4 + 16 * 1024 * 4); // 2x[16][1024]

  const int dir = blockIdx.x >> 2;
  const int b0g = (blockIdx.x & 3) * 16;
  const float* xproj = dir ? xprojB : xprojF;
  const f16*   Whh   = dir ? WhhBh  : WhhFh;
  float*       hseq  = dir ? hseqB  : hseqF;

  const int t = threadIdx.x;
  const int w = t >> 5, lane = t & 31;
  const int row16 = lane & 15;
  const int kg = (lane >> 4) * 8;

  for (int i = t; i < 16 * 256; i += 256) { hs[i] = (f16)0.f; cs[i] = 0.f; }
  if (w == 0)   // TDM prefetch of step 0's xproj slice
    tdm_load_16rows(xproj + (long long)b0g * S_ * H4, xbuf);
  __syncthreads();

  for (int step = 0; step < S_; ++step) {
    // kick DMA for step+1 while we compute step's gates
    if (w == 0 && step + 1 < S_)
      tdm_load_16rows(xproj + ((long long)b0g * S_ + (step + 1)) * H4,
                      xbuf + ((step + 1) & 1) * XBUF_ELEMS);

    // gates[16,1024] = h[16,256] @ Whh[1024,256]^T
    for (int nt = w; nt < 64; nt += 8) {
      v8f acc = {};
#pragma unroll
      for (int kk = 0; kk < 256; kk += 32) {
        v16h a  = load_frag(&hs[row16 * 256 + kk], kg);
        v16h bf = load_frag(&Whh[(long long)(nt * 16 + row16) * 256 + kk], kg);
        acc = __builtin_amdgcn_wmma_f32_16x16x32_f16(false, a, false, bf, (short)0, acc, false, false);
      }
#pragma unroll
      for (int r = 0; r < 8; ++r) {
        int m = r + ((lane >> 4) << 3);
        gs[m * 1024 + nt * 16 + row16] = acc[r];
      }
    }
    // ensure this step's TDM slice has landed (allow next step's to fly)
    if (w == 0) {
      if (step + 1 < S_) __builtin_amdgcn_s_wait_tensorcnt(1);
      else               __builtin_amdgcn_s_wait_tensorcnt(0);
    }
    __syncthreads();

    // activations: thread t owns hidden column j=t for all 16 rows
    const float* xb = xbuf + (step & 1) * XBUF_ELEMS;
    for (int m = 0; m < 16; ++m) {
      float gi = gs[m * 1024 + t]       + xb[m * 1024 + t];
      float gf = gs[m * 1024 + 256 + t] + xb[m * 1024 + 256 + t];
      float gg = gs[m * 1024 + 512 + t] + xb[m * 1024 + 512 + t];
      float go = gs[m * 1024 + 768 + t] + xb[m * 1024 + 768 + t];
      float c = sigf(gf) * cs[m * 256 + t] + sigf(gi) * tanhf(gg);
      float h = sigf(go) * tanhf(c);
      cs[m * 256 + t] = c;
      hs[m * 256 + t] = (f16)h;
      hseq[((long long)(b0g + m) * S_ + step) * H_ + t] = h;
    }
    __syncthreads();
  }
}

// ===================== small helper kernels =====================
__global__ void convert_pad_kernel(const float* __restrict__ src, f16* __restrict__ dst,
                                   int rows, int sk, int dk) {
  long long tid = (long long)blockIdx.x * 256 + threadIdx.x;
  if (tid >= (long long)rows * dk) return;
  int k = (int)(tid % dk); long long r = tid / dk;
  dst[tid] = (k < sk) ? (f16)src[r * sk + k] : (f16)0.f;
}

__global__ void comb_bias_kernel(const float* a, const float* b, float* c, int n) {
  int i = blockIdx.x * 256 + threadIdx.x;
  if (i < n) c[i] = a[i] + b[i];
}

__global__ void embed_kernel(const int* __restrict__ tok, const int* __restrict__ pos,
                             const int* __restrict__ post,
                             const float* __restrict__ emb, const float* __restrict__ pose,
                             const float* __restrict__ poste, f16* __restrict__ out) {
  long long tid = (long long)blockIdx.x * 256 + threadIdx.x;
  int k = (int)(tid % DINP); long long bs = tid / DINP;
  float v;
  if (k < 300)       v = emb[(long long)tok[bs] * 300 + k];
  else if (k < 330)  v = pose[(long long)pos[bs] * 30 + (k - 300)];
  else if (k < 360)  v = poste[(long long)post[bs] * 30 + (k - 330)];
  else               v = 0.f;
  out[tid] = (f16)v;
}

__global__ void gather_xb_kernel(const f16* __restrict__ embs, const int* __restrict__ tok,
                                 const int* __restrict__ len, f16* __restrict__ xb) {
  long long tid = (long long)blockIdx.x * 256 + threadIdx.x;
  int k = (int)(tid % DINP); long long bs = tid / DINP;
  int b = (int)(bs / S_), s = (int)(bs % S_);
  bool valid = tok[bs] != 0;
  int rev = valid ? (len[b] - 1 - s) : 0;
  xb[tid] = valid ? embs[((long long)b * S_ + rev) * DINP + k] : (f16)0.f;
}

__global__ void g_build_kernel(const float* __restrict__ hseqF, const float* __restrict__ hseqB,
                               const int* __restrict__ tok, const int* __restrict__ len,
                               float* __restrict__ gF, f16* __restrict__ gH) {
  long long tid = (long long)blockIdx.x * 256 + threadIdx.x;
  int j = (int)(tid % D_); long long bs = tid / D_;
  int b = (int)(bs / S_), s = (int)(bs % S_);
  bool valid = tok[bs] != 0;
  float v;
  if (j < H_) v = hseqF[bs * H_ + j];
  else {
    int rev = valid ? (len[b] - 1 - s) : 0;
    v = hseqB[((long long)b * S_ + rev) * H_ + (j - H_)];
  }
  v = valid ? v : 0.f;
  gF[tid] = v;
  gH[tid] = (f16)v;
}

// softmax over keys, mean over heads, diag=1, row mask
__global__ __launch_bounds__(256) void softmax_mean_kernel(
    const float* __restrict__ scores, const int* __restrict__ tok,
    float* __restrict__ adj_ag) {
  __shared__ float sbuf[8];
  int bq = blockIdx.x;
  int b = bq >> 9, q = bq & 511;
  int t = threadIdx.x;
  const int* tokrow = tok + (long long)b * S_;
  bool v0 = tokrow[t] != 0, v1 = tokrow[t + 256] != 0, vq = tokrow[q] != 0;
  const float scale = 0.08838834764831845f; // 1/sqrt(128)
  float a0 = 0.f, a1 = 0.f;
  for (int h = 0; h < NH_; ++h) {
    const float* row = scores + (((long long)(b * NH_ + h)) * S_ + q) * S_;
    float s0 = v0 ? row[t] * scale : -1e9f;
    float s1 = v1 ? row[t + 256] * scale : -1e9f;
    float m = blk_reduce(fmaxf(s0, s1), sbuf, 1);
    float e0 = __expf(s0 - m), e1 = __expf(s1 - m);
    float sum = blk_reduce(e0 + e1, sbuf, 0);
    float inv = 0.25f / sum;
    a0 += e0 * inv; a1 += e1 * inv;
  }
  long long ob = (long long)bq * S_;
  float o0 = (t == q) ? 1.f : a0;
  float o1 = (t + 256 == q) ? 1.f : a1;
  adj_ag[ob + t]       = vq ? o0 : 0.f;
  adj_ag[ob + t + 256] = vq ? o1 : 0.f;
}

// per-sample approximate k-th smallest nonzero via LDS histogram
__global__ __launch_bounds__(256) void prune_hist_kernel(
    const float* __restrict__ adj_ag, float* __restrict__ thre) {
  __shared__ int hist[NBIN];
  __shared__ int nnz_s;
  int b = blockIdx.x, t = threadIdx.x;
  for (int i = t; i < NBIN; i += 256) hist[i] = 0;
  if (t == 0) nnz_s = 0;
  __syncthreads();
  const float* base = adj_ag + (long long)b * S_ * S_;
  int ln = 0;
  for (int i = t; i < S_ * S_; i += 256) {
    float v = base[i];
    if (v != 0.f) {
      ++ln;
      int bin = (int)(v * NBIN); if (bin > NBIN - 1) bin = NBIN - 1;
      atomicAdd(&hist[bin], 1);
    }
  }
  atomicAdd(&nnz_s, ln);
  __syncthreads();
  if (t == 0) {
    int nnz = nnz_s;
    int kk = (int)floorf((float)(nnz - S_) * 0.3f);
    float th = 0.f;
    if (kk > 0) {
      int c = 0;
      for (int i = 0; i < NBIN; ++i) {
        c += hist[i];
        if (c >= kk) { th = (float)(i + 1) / (float)NBIN; break; }
      }
    }
    thre[b] = th;
  }
}

__global__ void prune_apply_kernel(float* __restrict__ adj_ag, const float* __restrict__ thre) {
  long long tid = (long long)blockIdx.x * 256 + threadIdx.x;
  int b = (int)(tid / ((long long)S_ * S_));
  float v = adj_ag[tid];
  adj_ag[tid] = (v <= thre[b]) ? 0.f : v;
}

__global__ __launch_bounds__(256) void ensemble_kernel(
    const float* __restrict__ adj, const float* __restrict__ adj_ag,
    f16* __restrict__ aens, float* __restrict__ denom) {
  __shared__ float sbuf[8];
  int row = blockIdx.x;                     // b*S + s
  long long base = (long long)row * S_;
  int t = threadIdx.x;
  float s = 0.f;
  for (int k = t; k < S_; k += 256) {
    float v = 0.5f * adj[base + k] + 0.5f * adj_ag[base + k];
    aens[base + k] = (f16)v;
    s += v;
  }
  float tot = blk_reduce(s, sbuf, 0);
  if (t == 0) denom[row] = tot + 1.f;
}

// ===================== host-side orchestration =====================
extern "C" void kernel_launch(void* const* d_in, const int* in_sizes, int n_in,
                              void* d_out, int out_size, void* d_ws, size_t ws_size,
                              hipStream_t stream) {
  (void)in_sizes; (void)n_in; (void)out_size; (void)ws_size;
  const float* adj   = (const float*)d_in[0];
  const int*   tok   = (const int*)d_in[1];
  const int*   pos   = (const int*)d_in[2];
  const int*   post  = (const int*)d_in[3];
  const int*   len   = (const int*)d_in[4];
  const float* emb   = (const float*)d_in[5];
  const float* pose  = (const float*)d_in[6];
  const float* poste = (const float*)d_in[7];
  const float* WihF  = (const float*)d_in[8];
  const float* WhhF  = (const float*)d_in[9];
  const float* bihF  = (const float*)d_in[10];
  const float* bhhF  = (const float*)d_in[11];
  const float* WihB  = (const float*)d_in[12];
  const float* WhhB  = (const float*)d_in[13];
  const float* bihB  = (const float*)d_in[14];
  const float* bhhB  = (const float*)d_in[15];
  const float* Wq    = (const float*)d_in[16];
  const float* bq    = (const float*)d_in[17];
  const float* Wk    = (const float*)d_in[18];
  const float* bk    = (const float*)d_in[19];
  const float* W0    = (const float*)d_in[20];
  const float* b0    = (const float*)d_in[21];
  const float* W1    = (const float*)d_in[22];
  const float* b1    = (const float*)d_in[23];

  float* outX   = (float*)d_out;                          // [B,S,MEM]
  float* outAdj = outX   + (size_t)B_ * S_ * MEM_;        // [B,S,S]
  float* outG   = outAdj + (size_t)B_ * S_ * S_;          // [B,S,2H]

  char* p = (char*)d_ws;
  auto alloc = [&](size_t bytes) -> void* {
    void* r = (void*)p; p += (bytes + 255) & ~(size_t)255; return r;
  };
  const size_t BS = (size_t)B_ * S_;
  f16*  embs_h = (f16*)alloc(BS * DINP * 2);
  f16*  xb_h   = (f16*)alloc(BS * DINP * 2);
  f16*  WihF_h = (f16*)alloc((size_t)H4 * DINP * 2);
  f16*  WihB_h = (f16*)alloc((size_t)H4 * DINP * 2);
  f16*  WhhF_h = (f16*)alloc((size_t)H4 * H_ * 2);
  f16*  WhhB_h = (f16*)alloc((size_t)H4 * H_ * 2);
  f16*  Wq_h   = (f16*)alloc((size_t)D_ * D_ * 2);
  f16*  Wk_h   = (f16*)alloc((size_t)D_ * D_ * 2);
  f16*  W0_h   = (f16*)alloc((size_t)MEM_ * D_ * 2);
  f16*  W1_h   = (f16*)alloc((size_t)MEM_ * MEM_ * 2);
  f16*  g_h    = (f16*)alloc(BS * D_ * 2);
  f16*  q_h    = (f16*)alloc(BS * D_ * 2);
  f16*  k_h    = (f16*)alloc(BS * D_ * 2);
  f16*  aens_h = (f16*)alloc(BS * S_ * 2);
  f16*  t1_h   = (f16*)alloc(BS * D_ * 2);
  f16*  x1_h   = (f16*)alloc(BS * MEM_ * 2);
  f16*  t3_h   = (f16*)alloc(BS * MEM_ * 2);
  float* combF  = (float*)alloc(H4 * 4);
  float* combB  = (float*)alloc(H4 * 4);
  float* xprojF = (float*)alloc(BS * H4 * 4);
  float* xprojB = (float*)alloc(BS * H4 * 4);
  float* hseqF  = (float*)alloc(BS * H_ * 4);
  float* hseqB  = (float*)alloc(BS * H_ * 4);
  float* scores = (float*)alloc((size_t)B_ * NH_ * S_ * S_ * 4);
  float* thre   = (float*)alloc(B_ * 4);
  float* denom  = (float*)alloc(BS * 4);

  // weight conversions (f32 -> f16, K padded where needed)
  auto cvt = [&](const float* s, f16* d, int rows, int sk, int dk) {
    long long tot = (long long)rows * dk;
    convert_pad_kernel<<<(unsigned)((tot + 255) / 256), 256, 0, stream>>>(s, d, rows, sk, dk);
  };
  cvt(WihF, WihF_h, H4, 360, DINP);
  cvt(WihB, WihB_h, H4, 360, DINP);
  cvt(WhhF, WhhF_h, H4, H_, H_);
  cvt(WhhB, WhhB_h, H4, H_, H_);
  cvt(Wq,   Wq_h,   D_, D_, D_);
  cvt(Wk,   Wk_h,   D_, D_, D_);
  cvt(W0,   W0_h,   MEM_, D_, D_);
  cvt(W1,   W1_h,   MEM_, MEM_, MEM_);
  comb_bias_kernel<<<4, 256, 0, stream>>>(bihF, bhhF, combF, H4);
  comb_bias_kernel<<<4, 256, 0, stream>>>(bihB, bhhB, combB, H4);

  // embeddings -> f16 (padded to 384)
  embed_kernel<<<(unsigned)(BS * DINP / 256), 256, 0, stream>>>(tok, pos, post, emb, pose, poste, embs_h);
  gather_xb_kernel<<<(unsigned)(BS * DINP / 256), 256, 0, stream>>>(embs_h, tok, len, xb_h);

  const int M = (int)BS;  // 32768
  auto gemm = [&](const f16* A, const f16* Bmat, float* Cf, f16* Ch,
                  const float* bias, const float* den,
                  int Mm, int Nn, int Kk, int lda, int ldb,
                  int Z, int zdiv, long long sA1, long long sA2,
                  long long sB1, long long sB2, long long sC1, long long sC2,
                  int bTrans, int relu) {
    dim3 grid(Nn / 64, Mm / 64, Z);
    gemm_wmma_kernel<<<grid, 256, 0, stream>>>(A, Bmat, Cf, Ch, bias, den,
        Mm, Nn, Kk, lda, ldb, zdiv, sA1, sA2, sB1, sB2, sC1, sC2, bTrans, relu);
  };

  // LSTM input projections: xproj = embs @ Wih^T + (bih+bhh)
  gemm(embs_h, WihF_h, xprojF, nullptr, combF, nullptr, M, H4, DINP, DINP, DINP,
       1, 1, 0, 0, 0, 0, 0, 0, 1, 0);
  gemm(xb_h,   WihB_h, xprojB, nullptr, combB, nullptr, M, H4, DINP, DINP, DINP,
       1, 1, 0, 0, 0, 0, 0, 0, 1, 0);

  // persistent LSTM recurrence (2 dirs x 4 batch groups), TDM-prefetched xproj
  lstm_rec_kernel<<<8, 256, LSTM_SMEM, stream>>>(xprojF, xprojB, WhhF_h, WhhB_h, hseqF, hseqB);

  // g = [out_f | gathered out_b] * vf  (f32 to output + f16 mirror)
  g_build_kernel<<<(unsigned)(BS * D_ / 256), 256, 0, stream>>>(hseqF, hseqB, tok, len, outG, g_h);

  // Q/K projections (f16 outputs only)
  gemm(g_h, Wq_h, nullptr, q_h, bq, nullptr, M, D_, D_, D_, D_, 1, 1, 0,0,0,0,0,0, 1, 0);
  gemm(g_h, Wk_h, nullptr, k_h, bk, nullptr, M, D_, D_, D_, D_, 1, 1, 0,0,0,0,0,0, 1, 0);

  // scores[b,h] = Q[b,:,h,:] @ K[b,:,h,:]^T ; z = b*NH+h
  gemm(q_h, k_h, scores, nullptr, nullptr, nullptr, S_, S_, 128, D_, D_,
       B_ * NH_, NH_,
       (long long)S_ * D_, 128, (long long)S_ * D_, 128,
       (long long)NH_ * S_ * S_, (long long)S_ * S_, 1, 0);

  // softmax over keys + head mean + diag/vf -> adj_ag (in d_out)
  softmax_mean_kernel<<<(unsigned)BS, 256, 0, stream>>>(scores, tok, outAdj);

  // dynamic-k prune (histogram threshold, in place)
  prune_hist_kernel<<<B_, 256, 0, stream>>>(outAdj, thre);
  prune_apply_kernel<<<(unsigned)(BS * S_ / 256), 256, 0, stream>>>(outAdj, thre);

  // adj_ens (f16) + denom = rowsum + 1
  ensemble_kernel<<<(unsigned)BS, 256, 0, stream>>>(adj, outAdj, aens_h, denom);

  // GCN layer 1: t1 = adj_ens @ g ; x1 = relu((t1 @ W0^T + b0)/denom)
  gemm(aens_h, g_h, nullptr, t1_h, nullptr, nullptr, S_, D_, S_, S_, D_,
       B_, 1, (long long)S_ * S_, 0, (long long)S_ * D_, 0, (long long)S_ * D_, 0, 0, 0);
  gemm(t1_h, W0_h, nullptr, x1_h, b0, denom, M, MEM_, D_, D_, D_,
       1, 1, 0,0,0,0,0,0, 1, 1);

  // GCN layer 2: t3 = adj_ens @ x1 ; x = relu((t3 @ W1^T + b1)/denom) -> d_out
  gemm(aens_h, x1_h, nullptr, t3_h, nullptr, nullptr, S_, MEM_, S_, S_, MEM_,
       B_, 1, (long long)S_ * S_, 0, (long long)S_ * MEM_, 0, (long long)S_ * MEM_, 0, 0, 0);
  gemm(t3_h, W1_h, outX, nullptr, b1, denom, M, MEM_, MEM_, MEM_, MEM_,
       1, 1, 0,0,0,0,0,0, 1, 1);
}
